// GCRNN_epsilon_early_stop_19473381720850
// MI455X (gfx1250) — matrix operand order repaired and checked by reading, not compiled, not run on hardware
//
#include <hip/hip_runtime.h>
#include <math.h>

// ---------------- types ----------------
typedef __attribute__((ext_vector_type(16))) __bf16 v16bf;
typedef __attribute__((ext_vector_type(8)))  float  v8f;

#define THREADS 256
#define WAVES   8

__device__ __forceinline__ unsigned short f2bf(float f) {
    union { float f; unsigned u; } x; x.f = f;
    unsigned r = x.u + 0x7FFFu + ((x.u >> 16) & 1u);   // RNE
    return (unsigned short)(r >> 16);
}

// ---- CDNA5 async global->LDS copy (ASYNCcnt path) ----
__device__ __forceinline__ void async_copy_b128(unsigned lds_off, const void* g) {
    asm volatile("global_load_async_to_lds_b128 %0, %1, off"
                 :: "v"(lds_off), "v"(g) : "memory");
}
__device__ __forceinline__ void wait_async_le2() {   // keep next panel's 2 in flight
    asm volatile("s_wait_asynccnt 0x2" ::: "memory");
}
__device__ __forceinline__ void wait_async_le0() {
    asm volatile("s_wait_asynccnt 0x0" ::: "memory");
}

// ---- wave32 butterfly reductions ----
__device__ __forceinline__ float wred_sum(float v) {
    #pragma unroll
    for (int m = 16; m; m >>= 1) v += __shfl_xor(v, m, 32);
    return v;
}
__device__ __forceinline__ float wred_max(float v) {
    #pragma unroll
    for (int m = 16; m; m >>= 1) v = fmaxf(v, __shfl_xor(v, m, 32));
    return v;
}

// ---------------- generic WMMA GEMM ----------------
// C[M x 128] = A[M x K] * B[K x 128]; fp32 accumulate via v_wmma_f32_16x16x32_bf16.
// A: lambda returning bf16 bits. B: pre-converted bf16, contiguous k-major [K][128];
// DOUBLE-BUFFERED async global->LDS staging: panel kp+32 is prefetched before waiting
// on panel kp (async ops complete in order; 2 per wave per panel -> wait asynccnt<=2).
// K multiple of 32 with an even panel count (128/448/768 all qualify). M<=256, mult of 16.
template <class ALoad, class Epi>
__device__ void gemm_n128(int M, int K, ALoad aload,
                          const unsigned short* __restrict__ Bsrc, Epi epi,
                          unsigned short* __restrict__ Apan,   // >= 256*32
                          unsigned short* __restrict__ Bpan) { // 2 x 32*128
    const int tid  = threadIdx.x;
    const int wave = tid >> 5;
    const int lane = tid & 31;
    const int nrb  = M >> 4;
    v8f acc[2][8];
    #pragma unroll
    for (int i = 0; i < 2; ++i)
        #pragma unroll
        for (int j = 0; j < 8; ++j)
            acc[i][j] = (v8f){0.f,0.f,0.f,0.f,0.f,0.f,0.f,0.f};

    const int kbase = (lane >> 4) * 8;       // ISA 16-bit A-matrix layout
    const int arow  = (lane & 15);
    const unsigned bofs0 = (unsigned)(unsigned long long)Bpan;
    const unsigned bofs1 = bofs0 + 32 * 128 * 2;

    // prologue: prefetch panel 0 into buf0 (buf0 is free: every GEMM ends on parity 1)
    {
        const unsigned short* gB = Bsrc;
        async_copy_b128(bofs0 + tid * 16, gB + tid * 8);
        async_copy_b128(bofs0 + (tid + THREADS) * 16, gB + (tid + THREADS) * 8);
    }

    for (int kp = 0; kp < K; kp += 32) {
        const unsigned cb = ((kp >> 5) & 1) ? bofs1 : bofs0;
        const unsigned nb = ((kp >> 5) & 1) ? bofs0 : bofs1;
        __syncthreads();                     // compute(kp-32) done: Apan + other buf free
        const bool more = (kp + 32) < K;
        if (more) {                          // prefetch next panel into the other buffer
            const unsigned short* gB = Bsrc + (kp + 32) * 128;
            async_copy_b128(nb + tid * 16, gB + tid * 8);
            async_copy_b128(nb + (tid + THREADS) * 16, gB + (tid + THREADS) * 8);
        }
        // --- A panel (VALU convert/copy), M x 32; overlaps in-flight DMA ---
        for (int idx = tid; idx < M * 32; idx += THREADS) {
            int r = idx >> 5, kk = idx & 31;
            Apan[(r << 5) + kk] = aload(r, kp + kk);
        }
        if (more) wait_async_le2(); else wait_async_le0();
        __syncthreads();

        const unsigned short* Bcur = (const unsigned short*)(Bpan + (((kp >> 5) & 1) ? 4096 : 0));
        union BF { unsigned short u[16]; v16bf v; } bf[8];
        #pragma unroll
        for (int ct = 0; ct < 8; ++ct)
            #pragma unroll
            for (int e = 0; e < 16; ++e)
                bf[ct].u[e] = Bcur[((kbase + (e & 7) + ((e >> 3) << 4)) << 7) + (ct << 4) + (lane & 15)];

        for (int rb = wave, ri = 0; rb < nrb; rb += WAVES, ++ri) {
            union AF { unsigned short u[16]; v16bf v; } af;
            const unsigned short* ap = &Apan[((rb << 4) + arow) << 5];
            #pragma unroll
            for (int e = 0; e < 8; ++e) { af.u[e] = ap[kbase + e]; af.u[8 + e] = ap[16 + kbase + e]; }
            #pragma unroll
            for (int ct = 0; ct < 8; ++ct)
                acc[ri][ct] = __builtin_amdgcn_wmma_f32_16x16x32_bf16(
                    false, af.v, false, bf[ct].v, (short)0, acc[ri][ct], false, false);
        }
        (void)cb;
    }
    __syncthreads();
    const int lr = lane & 15, lh = lane >> 4;  // C layout: M = r + 8*(lane>>4), N = lane&15
    for (int rb = wave, ri = 0; rb < nrb; rb += WAVES, ++ri)
        #pragma unroll
        for (int ct = 0; ct < 8; ++ct)
            #pragma unroll
            for (int r = 0; r < 8; ++r)
                epi((rb << 4) + r + (lh << 3), (ct << 4) + lr, acc[ri][ct][r]);
}

// ---------------- params ----------------
struct LayerP {
    const float *cg_W, *cg_b, *embed;
    const float *b1, *b2, *bb1, *bb2, *g1, *g2;
    const float *in_b, *in_w, *out_b, *out_w, *w1, *w2;
    const float *ug_W, *ug_b;
};
struct Params {
    const float *x, *adj, *dt, *mask;
    const int*  lengths;
    LayerP L[2];
    float* out;
    float* ws;
};

// ---- bf16 weight region (ushort elements), at ws offset 0 ----
#define WOFF_ADJT 0
#define WOFF_L0   16384
#define WOFF_L1   229376
#define WREG_US   524288
#define WREG_BYTES (WREG_US * 2)

// ---- per-block activation region (bytes) ----
#define PB_H0   0            // f32 128x128
#define PB_H1   65536
#define PB_OA   131072
#define PB_Q    196608       // bf16 [s][d]
#define PB_KT   229376       // bf16 [d][t]
#define PB_V    262144       // bf16 [t][d]
#define PB_O    294912       // bf16 [s][d]
#define PB_HID  327680       // bf16 [s][d]
#define PB_XO   360448       // bf16 [3*cinp][128] (768 rows max)
#define PB_BYTES 557056

// ---------------- one-time weight prep (f32 -> bf16, transpose, pad) ----------------
__global__ void prep_weights(Params P) {
    unsigned short* wr = (unsigned short*)P.ws;
    const int tid = blockIdx.x * blockDim.x + threadIdx.x;
    const int NT  = gridDim.x * blockDim.x;
    for (int i = tid; i < 16384; i += NT) {              // adjT[k][n] = adj[n][k]
        int k = i >> 7, n = i & 127;
        wr[WOFF_ADJT + i] = f2bf(P.adj[(n << 7) + k]);
    }
    for (int l = 0; l < 2; ++l) {
        const LayerP& L = P.L[l];
        unsigned short* base = wr + ((l == 0) ? WOFF_L0 : WOFF_L1);
        for (int i = tid; i < 49152; i += NT) {          // inT[p][k][n] = in_w[p*128+n][k]
            int p = i >> 14, r = i & 16383, k = r >> 7, n = r & 127;
            base[i] = f2bf(L.in_w[((p << 7) + n) * 128 + k]);
        }
        unsigned short* oT = base + 49152;
        for (int i = tid; i < 16384; i += NT) {
            int k = i >> 7, n = i & 127;
            oT[i]         = f2bf(L.out_w[(n << 7) + k]);
            oT[16384 + i] = f2bf(L.w1[(n << 7) + k]);
            oT[32768 + i] = f2bf(L.w2[(n << 7) + k]);
        }
        const int cin = (l == 0) ? 129 : 256;
        const int cinp = (l == 0) ? 144 : 256;
        const int KP  = (l == 0) ? 448 : 768;
        unsigned short* Wu = base + 49152 + 3 * 16384;
        unsigned short* Wc = Wu + 128 * KP;
        for (int i = tid; i < 128 * KP; i += NT) {
            int r = i / KP, k = i - r * KP;
            float vu = 0.f, vc = 0.f;
            if (k < 3 * cinp) {
                int o = k / cinp, c = k - o * cinp;
                if (c < cin) {
                    vu = L.ug_W[r * (3 * cin) + o * cin + c];
                    vc = L.cg_W[r * (3 * cin) + o * cin + c];
                }
            }
            Wu[i] = f2bf(vu); Wc[i] = f2bf(vc);
        }
    }
}

// ---------------- persistent per-batch kernel ----------------
__global__ __launch_bounds__(THREADS) void gcrnn_persistent(Params P) {
    extern __shared__ char smem[];
    float* ti  = (float*)smem;                             // 128x128 tokens / u-gate
    float* tmp = ti + 16384;                               // scores / ff / c-gate
    unsigned short* Apan = (unsigned short*)(tmp + 16384); // 256x32 bf16
    unsigned short* Bpan = Apan + 256 * 32;                // 2 x (32x128) bf16

    const int b    = blockIdx.x;
    const int tid  = threadIdx.x;
    const int wave = tid >> 5;
    const int lane = tid & 31;
    const unsigned short* wr = (const unsigned short*)P.ws;
    const unsigned short* adjT = wr + WOFF_ADJT;

    char* base = (char*)P.ws + WREG_BYTES + (size_t)b * PB_BYTES;
    float* H0 = (float*)(base + PB_H0);
    float* H1 = (float*)(base + PB_H1);
    float* OA = (float*)(base + PB_OA);
    unsigned short* Qb  = (unsigned short*)(base + PB_Q);
    unsigned short* KTb = (unsigned short*)(base + PB_KT);
    unsigned short* Vb  = (unsigned short*)(base + PB_V);
    unsigned short* Ob  = (unsigned short*)(base + PB_O);
    unsigned short* Hid = (unsigned short*)(base + PB_HID);
    unsigned short* XO  = (unsigned short*)(base + PB_XO);

    for (int i = tid; i < 16384; i += THREADS) { H0[i] = 0.f; H1[i] = 0.f; OA[i] = 0.f; }
    __syncthreads();

    const int len_b = P.lengths[b];

    for (int t = 0; t < 64; ++t) {
        const float dtv = P.dt[b * 64 + t];
        const float sq1 = sqrtf(1.0f - dtv), sqd = sqrtf(dtv);

        for (int l = 0; l < 2; ++l) {
            const LayerP& Lp = P.L[l];
            const unsigned short* LB  = wr + ((l == 0) ? WOFF_L0 : WOFF_L1);
            const unsigned short* inT = LB;
            const unsigned short* outT = LB + 49152;
            const unsigned short* w1T  = LB + 65536;
            const unsigned short* w2T  = LB + 81920;
            const int KP = (l == 0) ? 448 : 768;
            const unsigned short* Wu = LB + 98304;
            const unsigned short* Wc = Wu + 128 * KP;
            float* h = (l == 0) ? H0 : H1;

            if (tid == 0) { __builtin_prefetch(Wu, 0, 1); __builtin_prefetch(Wc, 0, 1); }

            // ---- ti[n][d] = h[d][n] + embed[n][d] * mask_last[n]
            {
                const int tm = (t > 0) ? (t - 1) : 0;
                for (int i = tid; i < 16384; i += THREADS) {
                    int n = i >> 7, d = i & 127;
                    float ml = P.mask[(b * 128 + n) * 64 + tm];
                    ti[i] = h[(d << 7) + n] + Lp.embed[(n << 7) + d] * ml;
                }
            }
            __syncthreads();

            // ---- q, k, v (epilogues emit bf16 in WMMA-friendly layouts)
            gemm_n128(128, 128,
                [&](int r, int k) { return f2bf(ti[(r << 7) + k]); }, inT,
                [&](int r, int c, float v) { Qb[(r << 7) + c] = f2bf(v + Lp.in_b[c]); },
                Apan, Bpan);
            gemm_n128(128, 128,
                [&](int r, int k) { return f2bf(ti[(r << 7) + k]); }, inT + 16384,
                [&](int r, int c, float v) { KTb[(c << 7) + r] = f2bf(v + Lp.in_b[128 + c]); },
                Apan, Bpan);
            gemm_n128(128, 128,
                [&](int r, int k) { return f2bf(ti[(r << 7) + k]); }, inT + 32768,
                [&](int r, int c, float v) { Vb[(r << 7) + c] = f2bf(v + Lp.in_b[256 + c]); },
                Apan, Bpan);
            // ---- scores = q kT / sqrt(d) -> tmp (f32)
            gemm_n128(128, 128,
                [&](int r, int k) { return Qb[(r << 7) + k]; }, KTb,
                [&](int r, int c, float v) { tmp[(r << 7) + c] = v * 0.08838834764831845f; },
                Apan, Bpan);
            __syncthreads();
            // ---- softmax: one wave per row, 4 elems/lane, shfl butterflies
            for (int r = wave; r < 128; r += WAVES) {
                float* row = tmp + (r << 7);
                float x0 = row[lane], x1 = row[lane + 32], x2 = row[lane + 64], x3 = row[lane + 96];
                float m = wred_max(fmaxf(fmaxf(x0, x1), fmaxf(x2, x3)));
                x0 = __expf(x0 - m); x1 = __expf(x1 - m); x2 = __expf(x2 - m); x3 = __expf(x3 - m);
                float inv = 1.0f / wred_sum(x0 + x1 + x2 + x3);
                row[lane] = x0 * inv; row[lane + 32] = x1 * inv;
                row[lane + 64] = x2 * inv; row[lane + 96] = x3 * inv;
            }
            __syncthreads();
            // ---- o = a @ v -> Ob (bf16)
            gemm_n128(128, 128,
                [&](int r, int k) { return f2bf(tmp[(r << 7) + k]); }, Vb,
                [&](int r, int c, float v) { Ob[(r << 7) + c] = f2bf(v); },
                Apan, Bpan);
            // ---- attn = o @ out_w.T + out_b -> tmp (f32)
            gemm_n128(128, 128,
                [&](int r, int k) { return Ob[(r << 7) + k]; }, outT,
                [&](int r, int c, float v) { tmp[(r << 7) + c] = v + Lp.out_b[c]; },
                Apan, Bpan);
            __syncthreads();
            // ---- x1 = LN(ti + attn; g1, b1) -> ti   (wave-parallel rows)
            for (int r = wave; r < 128; r += WAVES) {
                float* xr = ti + (r << 7); const float* ar = tmp + (r << 7);
                float v[4], s = 0.f, ss = 0.f;
                #pragma unroll
                for (int j = 0; j < 4; ++j) {
                    float u = xr[lane + 32 * j] + ar[lane + 32 * j];
                    v[j] = u; s += u; ss += u * u;
                }
                s = wred_sum(s); ss = wred_sum(ss);
                float mean = s * (1.0f / 128.0f);
                float inv = rsqrtf(ss * (1.0f / 128.0f) - mean * mean + 1e-5f);
                #pragma unroll
                for (int j = 0; j < 4; ++j) {
                    int c = lane + 32 * j;
                    xr[c] = (v[j] - mean) * inv * Lp.g1[c] + Lp.b1[c];
                }
            }
            __syncthreads();
            // ---- ff1 = relu(x1 @ w1.T + bb1) -> Hid (bf16)
            gemm_n128(128, 128,
                [&](int r, int k) { return f2bf(ti[(r << 7) + k]); }, w1T,
                [&](int r, int c, float v) { v += Lp.bb1[c]; Hid[(r << 7) + c] = f2bf(v > 0.f ? v : 0.f); },
                Apan, Bpan);
            // ---- ff2 -> tmp (f32)
            gemm_n128(128, 128,
                [&](int r, int k) { return Hid[(r << 7) + k]; }, w2T,
                [&](int r, int c, float v) { tmp[(r << 7) + c] = v + Lp.bb2[c]; },
                Apan, Bpan);
            __syncthreads();
            // ---- eps = tanh(LN(x1 + ff; g2, b2)) -> ti   (wave-parallel rows)
            for (int r = wave; r < 128; r += WAVES) {
                float* xr = ti + (r << 7); const float* fr = tmp + (r << 7);
                float v[4], s = 0.f, ss = 0.f;
                #pragma unroll
                for (int j = 0; j < 4; ++j) {
                    float u = xr[lane + 32 * j] + fr[lane + 32 * j];
                    v[j] = u; s += u; ss += u * u;
                }
                s = wred_sum(s); ss = wred_sum(ss);
                float mean = s * (1.0f / 128.0f);
                float inv = rsqrtf(ss * (1.0f / 128.0f) - mean * mean + 1e-5f);
                #pragma unroll
                for (int j = 0; j < 4; ++j) {
                    int c = lane + 32 * j;
                    xr[c] = tanhf((v[j] - mean) * inv * Lp.g2[c] + Lp.b2[c]);
                }
            }
            __syncthreads();

            // ---- h_corr + build xg order-0 rows of XO (bf16)
            const int cin_in = (l == 0) ? 1 : 128;
            const int cinp   = (l == 0) ? 144 : 256;
            if (l == 0) {
                for (int n = tid; n < 128; n += THREADS)
                    XO[n] = f2bf(P.x[(b * 128 + n) * 64 + t]);
                for (int i = 129 * 128 + tid; i < 144 * 128; i += THREADS) XO[i] = 0;
            } else {
                for (int i = tid; i < 16384; i += THREADS) XO[i] = f2bf(H0[i]);
            }
            for (int i = tid; i < 16384; i += THREADS) {
                float hv = h[i];
                float ev = ti[((i & 127) << 7) + (i >> 7)];
                float hc = (t == 0) ? hv : (sq1 * hv - sqd * ev);
                h[i] = hc;
                XO[(cin_in << 7) + i] = f2bf(hc);
            }
            __syncthreads();

            // ---- diffusion orders 1, 2 into XO (contiguous k-major for proj B)
            gemm_n128(cinp, 128,
                [&](int r, int k) { return XO[(r << 7) + k]; }, adjT,
                [&](int r, int c, float v) { XO[((cinp + r) << 7) + c] = f2bf(v); },
                Apan, Bpan);
            gemm_n128(cinp, 128,
                [&](int r, int k) { return XO[((cinp + r) << 7) + k]; }, adjT,
                [&](int r, int c, float v) { XO[((2 * cinp + r) << 7) + c] = f2bf(v); },
                Apan, Bpan);

            // ---- gates: u -> ti, c -> tmp (zero-padded Wu/Wc kill pad/garbage rows)
            gemm_n128(128, KP,
                [&](int r, int k) { return Wu[r * KP + k]; }, XO,
                [&](int r, int c, float v) { v += Lp.ug_b[r];
                                             ti[(r << 7) + c] = 1.0f / (1.0f + __expf(-v)); },
                Apan, Bpan);
            gemm_n128(128, KP,
                [&](int r, int k) { return Wc[r * KP + k]; }, XO,
                [&](int r, int c, float v) { tmp[(r << 7) + c] = tanhf(v + Lp.cg_b[r]); },
                Apan, Bpan);
            __syncthreads();
            // ---- h = u*h + (1-u)*c
            for (int i = tid; i < 16384; i += THREADS) {
                float u = ti[i], cc = tmp[i];
                h[i] = u * h[i] + (1.0f - u) * cc;
            }
            __syncthreads();
        } // layers

        if (len_b - 1 == t)
            for (int i = tid; i < 16384; i += THREADS) OA[i] = H1[i];
    } // steps

    for (int i = tid; i < 16384; i += THREADS)
        P.out[(size_t)b * 16384 + i] = OA[i];
}

// ---------------- host launcher ----------------
extern "C" void kernel_launch(void* const* d_in, const int* in_sizes, int n_in,
                              void* d_out, int out_size, void* d_ws, size_t ws_size,
                              hipStream_t stream) {
    (void)in_sizes; (void)n_in; (void)out_size; (void)ws_size;
    Params P;
    P.x    = (const float*)d_in[0];
    P.adj  = (const float*)d_in[1];
    P.dt   = (const float*)d_in[2];
    /* d_in[3] = time_encoding (unused by the reference) */
    P.mask = (const float*)d_in[4];
    int i = 5;
    for (int l = 0; l < 2; ++l) {
        LayerP& L = P.L[l];
        // pytree flatten order: sorted dict keys: cg < embed < tr < ug
        L.cg_W  = (const float*)d_in[i++];
        L.cg_b  = (const float*)d_in[i++];
        L.embed = (const float*)d_in[i++];
        // tr keys sorted: b1, b2, bb1, bb2, g1, g2, in_b, in_w, out_b, out_w, w1, w2
        L.b1    = (const float*)d_in[i++];
        L.b2    = (const float*)d_in[i++];
        L.bb1   = (const float*)d_in[i++];
        L.bb2   = (const float*)d_in[i++];
        L.g1    = (const float*)d_in[i++];
        L.g2    = (const float*)d_in[i++];
        L.in_b  = (const float*)d_in[i++];
        L.in_w  = (const float*)d_in[i++];
        L.out_b = (const float*)d_in[i++];
        L.out_w = (const float*)d_in[i++];
        L.w1    = (const float*)d_in[i++];
        L.w2    = (const float*)d_in[i++];
        L.ug_W  = (const float*)d_in[i++];
        L.ug_b  = (const float*)d_in[i++];
    }
    P.lengths = (const int*)d_in[i];
    P.out = (float*)d_out;
    P.ws  = (float*)d_ws;

    const size_t shmem = (size_t)(16384 + 16384) * sizeof(float)        // ti + tmp
                       + (size_t)(256 * 32) * sizeof(unsigned short)    // A panel
                       + (size_t)(2 * 32 * 128) * sizeof(unsigned short); // B panels x2
    (void)hipFuncSetAttribute((const void*)gcrnn_persistent,
                              hipFuncAttributeMaxDynamicSharedMemorySize, (int)shmem);

    hipLaunchKernelGGL(prep_weights, dim3(256), dim3(THREADS), 0, stream, P);
    hipLaunchKernelGGL(gcrnn_persistent, dim3(32), dim3(THREADS), shmem, stream, P);
}
